// NewAttention_34943853920599
// MI455X (gfx1250) — matrix operand order repaired and checked by reading, not compiled, not run on hardware
//
#include <hip/hip_runtime.h>
#include <hip/hip_bf16.h>
#include <math.h>
#include <stdint.h>

// ---------------------------------------------------------------------------
// CDNA5 (gfx1250) types for WMMA 16x16x32 bf16 (wave32)
// ---------------------------------------------------------------------------
typedef __attribute__((ext_vector_type(16))) __bf16 v16bf;
typedef __attribute__((ext_vector_type(8)))  __bf16 v8bf;
typedef __attribute__((ext_vector_type(8)))  float  v8f;
typedef __attribute__((ext_vector_type(4)))  unsigned int uintv4;
typedef __attribute__((ext_vector_type(4)))  int intv4;
typedef __attribute__((ext_vector_type(8)))  int intv8;

union V16 { v16bf v; v8bf h[2]; };

static __device__ __forceinline__ unsigned short f2bf_bits(float f) {
  union { float f; unsigned int u; } v; v.f = f;
  unsigned int u = v.u;
  return (unsigned short)((u + 0x7FFFu + ((u >> 16) & 1u)) >> 16);  // RNE
}
static __device__ __forceinline__ float bf2f(unsigned short s) {
  union { unsigned int u; float f; } v; v.u = ((unsigned int)s) << 16;
  return v.f;
}

static __device__ __forceinline__ v8f wmma_bf16(v16bf a, v16bf b, v8f c) {
  return __builtin_amdgcn_wmma_f32_16x16x32_bf16(false, a, false, b,
                                                 (short)0, c, false, false);
}

// ---------------------------------------------------------------------------
// Tensor Data Mover: 2-D bf16 tile (tile_w x tile_h elements) global -> LDS.
// D# per CDNA5 ISA ch.8: group0 {count=1, lds_addr, global_addr, type=2},
// group1 {data_size=2B, tensor dims, tile dims, dim0 stride}.
// ---------------------------------------------------------------------------
static __device__ __forceinline__ void tdm_load_2d_bf16(
    unsigned lds_addr, const void* gptr,
    unsigned tile_w, unsigned tile_h,
    unsigned long long row_stride_elems,
    unsigned tensor_w, unsigned tensor_h) {
  unsigned long long ga = (unsigned long long)(uintptr_t)gptr;
  uintv4 g0;
  g0.x = 1u;                                                 // count=1 (valid)
  g0.y = lds_addr;                                           // LDS byte addr
  g0.z = (unsigned)(ga & 0xFFFFFFFFu);                       // global addr lo
  g0.w = (unsigned)((ga >> 32) & 0x01FFFFFFu) | (2u << 30);  // addr hi | type=2
  unsigned w0 = (1u << 16);                                  // data_size=1 (2B)
  unsigned w1 = (tensor_w & 0xFFFFu) << 16;                  // tensor_dim0 lo
  unsigned w2 = (tensor_w >> 16) | ((tensor_h & 0xFFFFu) << 16);
  unsigned w3 = (tensor_h >> 16) | (tile_w << 16);           // tile_dim0
  unsigned w4 = tile_h;                                      // tile_dim1 (dim2=0)
  unsigned w5 = (unsigned)(row_stride_elems & 0xFFFFFFFFull);
  unsigned w6 = (unsigned)((row_stride_elems >> 32) & 0xFFFFull);
  intv8 g1 = {(int)w0, (int)w1, (int)w2, (int)w3,
              (int)w4, (int)w5, (int)w6, 0};
  intv4 z4 = {0, 0, 0, 0};
#if __clang_major__ >= 23
  intv8 z8 = {0, 0, 0, 0, 0, 0, 0, 0};
  __builtin_amdgcn_tensor_load_to_lds(g0, g1, z4, z4, z8, 0);
#else
  __builtin_amdgcn_tensor_load_to_lds(g0, g1, z4, z4, 0);
#endif
}

// ---------------------------------------------------------------------------
// fp32 -> bf16 conversion
// ---------------------------------------------------------------------------
__global__ void f32_to_bf16_kernel(const float* __restrict__ in,
                                   unsigned short* __restrict__ out, int n) {
  int i = blockIdx.x * blockDim.x + threadIdx.x;
  if (i < n) out[i] = f2bf_bits(in[i]);
}

// ---------------------------------------------------------------------------
// GEMM: C[4096,2048] = A[4096,2048](bf16) @ W[2048,2048](bf16)^T
// Block 256 thr = 8 waves; block tile 32(M) x 512(N); wave tile 32x64:
// two A fragments share four B fragments -> 8 WMMAs per 12 fragment loads.
// A tile (32x64) TDM-staged into LDS, double buffered; W streamed from global.
// ---------------------------------------------------------------------------
template <bool STORE_F32>
__global__ __launch_bounds__(256)
void gemm_xwT_kernel(const unsigned short* __restrict__ A,
                     const unsigned short* __restrict__ W,
                     unsigned short* __restrict__ Cb,
                     float* __restrict__ Cf) {
  constexpr int N = 2048, K = 2048;
  constexpr int KC = 64;                 // K-chunk per TDM fetch
  constexpr int NC = K / KC;             // 32 chunks
  constexpr int MT = 32;                 // block/wave M tile
  const int wave = threadIdx.x >> 5;
  const int lane = threadIdx.x & 31;
  const int lr   = lane & 15;
  const int hi   = lane >> 4;
  const int m0 = blockIdx.x * MT;
  const int n0 = blockIdx.y * 512 + wave * 64;

  __shared__ __attribute__((aligned(16))) unsigned short abuf[2][MT * KC];
  const __bf16* Wb = (const __bf16*)W;
  const unsigned short* atile = A + (size_t)m0 * K;

  if (wave == 0) {
    tdm_load_2d_bf16((unsigned)(uintptr_t)&abuf[0][0], atile,
                     KC, MT, K, K, 1u << 20);
  }

  v8f acc[2][4] = {};

  for (int c = 0; c < NC; ++c) {
    if (wave == 0) {
      if (c + 1 < NC) {
        tdm_load_2d_bf16((unsigned)(uintptr_t)&abuf[(c + 1) & 1][0],
                         atile + (c + 1) * KC, KC, MT, K, K, 1u << 20);
        __builtin_amdgcn_s_wait_tensorcnt(1);   // oldest fetch (chunk c) done
      } else {
        __builtin_amdgcn_s_wait_tensorcnt(0);
      }
    }
    __syncthreads();                            // chunk c visible to all waves
    const __bf16* ab = (const __bf16*)&abuf[c & 1][0];
#pragma unroll
    for (int s = 0; s < 2; ++s) {
      const int kk = c * KC + s * 32;
      V16 af[2];
#pragma unroll
      for (int r = 0; r < 2; ++r) {
        const __bf16* arow = ab + (r * 16 + lr) * KC + s * 32 + hi * 8;
        af[r].h[0] = *(const v8bf*)(arow);
        af[r].h[1] = *(const v8bf*)(arow + 16);
      }
      V16 bfr[4];
#pragma unroll
      for (int t = 0; t < 4; ++t) {
        const __bf16* wrow = Wb + (size_t)(n0 + t * 16 + lr) * K + kk + hi * 8;
        __builtin_prefetch((const void*)(wrow + KC), 0, 0);
        bfr[t].h[0] = *(const v8bf*)(wrow);
        bfr[t].h[1] = *(const v8bf*)(wrow + 16);
      }
#pragma unroll
      for (int t = 0; t < 4; ++t) {
        acc[0][t] = wmma_bf16(af[0].v, bfr[t].v, acc[0][t]);
        acc[1][t] = wmma_bf16(af[1].v, bfr[t].v, acc[1][t]);
      }
    }
    __syncthreads();                            // buffer consumed before re-fill
  }

  // C layout: element i, lane l -> row m0 + r*16 + i + 8*hi, col n0 + t*16 + lr
#pragma unroll
  for (int r = 0; r < 2; ++r) {
#pragma unroll
    for (int t = 0; t < 4; ++t) {
#pragma unroll
      for (int i = 0; i < 8; ++i) {
        const size_t m = (size_t)(m0 + r * 16 + i + 8 * hi);
        const size_t n = (size_t)(n0 + t * 16 + lr);
        if (STORE_F32) Cf[m * N + n] = acc[r][t][i];
        else           Cb[m * N + n] = f2bf_bits(acc[r][t][i]);
      }
    }
  }
}

// ---------------------------------------------------------------------------
// RoPE in-place on bf16 [B,S,H,Dh], theta=10000
// ---------------------------------------------------------------------------
__global__ void rope_bf16_kernel(unsigned short* __restrict__ t, int n) {
  constexpr int S = 2048, H = 16, Dh = 128, half = Dh / 2;
  int idx = blockIdx.x * blockDim.x + threadIdx.x;
  if (idx >= n) return;
  int i = idx & (half - 1);
  int h = (idx >> 6) & (H - 1);
  int s = (idx >> 10) & (S - 1);
  int b = idx >> 21;
  float inv = __powf(10000.0f, -(2.0f * (float)i) / (float)Dh);
  float ang = (float)s * inv;
  float sn, cs;
  __sincosf(ang, &sn, &cs);
  size_t base = (((size_t)b * S + s) * H + h) * Dh + 2 * i;
  float xr = bf2f(t[base]);
  float xi = bf2f(t[base + 1]);
  t[base]     = f2bf_bits(xr * cs - xi * sn);
  t[base + 1] = f2bf_bits(xr * sn + xi * cs);
}

// ---------------------------------------------------------------------------
// Transpose V: [B,S,H,Dh] -> [B,H,Dh,S]
// ---------------------------------------------------------------------------
__global__ void transpose_v_kernel(const unsigned short* __restrict__ vin,
                                   unsigned short* __restrict__ vout, int n) {
  constexpr int S = 2048, H = 16, Dh = 128;
  int idx = blockIdx.x * blockDim.x + threadIdx.x;
  if (idx >= n) return;
  int s  = idx & (S - 1);
  int dh = (idx >> 11) & (Dh - 1);
  int h  = (idx >> 18) & (H - 1);
  int b  = idx >> 22;
  vout[idx] = vin[(((size_t)b * S + s) * H + h) * Dh + dh];
}

// ---------------------------------------------------------------------------
// One 32-key tile of causal flash attention. MASKED=true only for the single
// diagonal tile; all earlier tiles are fully below the diagonal.
// ---------------------------------------------------------------------------
template <bool MASKED>
static __device__ __forceinline__ void fa_process_tile(
    const __bf16* __restrict__ ks, const __bf16* __restrict__ vtb,
    unsigned short* __restrict__ pb,
    const V16 (&qf)[4], v8f (&o)[8], float (&mrow)[8], float (&lrow)[8],
    const int k0, const int qt, const int lr, const int hi) {
  constexpr int Dh = 128, S = 2048;
  const float scale = 0.08838834764831845f;     // 1/sqrt(128)

  // ---- S = Q @ K^T: 8 B-fragments from LDS, then 8 WMMAs ----
  V16 bfr[8];
#pragma unroll
  for (int t = 0; t < 4; ++t) {
    const __bf16* kr0 = ks + lr * Dh + t * 32 + hi * 8;        // keys k0+lr
    const __bf16* kr1 = ks + (16 + lr) * Dh + t * 32 + hi * 8; // keys k0+16+lr
    bfr[t].h[0]     = *(const v8bf*)(kr0);
    bfr[t].h[1]     = *(const v8bf*)(kr0 + 16);
    bfr[4 + t].h[0] = *(const v8bf*)(kr1);
    bfr[4 + t].h[1] = *(const v8bf*)(kr1 + 16);
  }
  v8f s0 = {}, s1 = {};
#pragma unroll
  for (int t = 0; t < 4; ++t) {
    s0 = wmma_bf16(qf[t].v, bfr[t].v, s0);
    s1 = wmma_bf16(qf[t].v, bfr[4 + t].v, s1);
  }

  // ---- (mask +) online softmax ----
  float p0[8], p1[8];
#pragma unroll
  for (int i = 0; i < 8; ++i) {
    float a0 = s0[i] * scale;
    float a1 = s1[i] * scale;
    if (MASKED) {
      const int qrow = qt * 16 + i + 8 * hi;
      a0 = (k0 + lr      <= qrow) ? a0 : -1e30f;
      a1 = (k0 + 16 + lr <= qrow) ? a1 : -1e30f;
    }
    float mx = fmaxf(a0, a1);
#pragma unroll
    for (int d = 1; d < 16; d <<= 1) mx = fmaxf(mx, __shfl_xor(mx, d, 16));
    const float mnew = fmaxf(mrow[i], mx);
    const float corr = __expf(mrow[i] - mnew);
    mrow[i] = mnew;
    const float e0 = __expf(a0 - mnew);
    const float e1 = __expf(a1 - mnew);
    float rs = e0 + e1;
#pragma unroll
    for (int d = 1; d < 16; d <<= 1) rs += __shfl_xor(rs, d, 16);
    lrow[i] = lrow[i] * corr + rs;
    p0[i] = e0; p1[i] = e1;
#pragma unroll
    for (int t = 0; t < 8; ++t) o[t][i] *= corr;
  }

  // ---- re-layout P (C layout) -> A layout via LDS ----
#pragma unroll
  for (int i = 0; i < 8; ++i) {
    const int row = i + 8 * hi;
    pb[row * 32 + lr]      = f2bf_bits(p0[i]);
    pb[row * 32 + 16 + lr] = f2bf_bits(p1[i]);
  }
  V16 pf;
  pf.h[0] = *(const v8bf*)((const __bf16*)pb + lr * 32 + hi * 8);
  pf.h[1] = *(const v8bf*)((const __bf16*)pb + lr * 32 + 16 + hi * 8);

  // ---- O += P @ V: 8 V-fragments (global, contiguous), then 8 WMMAs ----
  V16 vf[8];
#pragma unroll
  for (int nt = 0; nt < 8; ++nt) {
    const __bf16* vrow = vtb + (size_t)(nt * 16 + lr) * S + k0 + hi * 8;
    vf[nt].h[0] = *(const v8bf*)(vrow);
    vf[nt].h[1] = *(const v8bf*)(vrow + 16);
  }
#pragma unroll
  for (int nt = 0; nt < 8; ++nt) o[nt] = wmma_bf16(pf.v, vf[nt].v, o[nt]);
}

// ---------------------------------------------------------------------------
// Flash attention (causal). One wave = one (b, h, 16-query tile).
// K-tiles (32 keys x 128 dh) TDM-staged per wave into LDS, double buffered
// (wave-private: no barriers, just s_wait_tensorcnt).
// ---------------------------------------------------------------------------
#define FA_WAVES 2
__global__ __launch_bounds__(FA_WAVES * 32)
void flash_attn_kernel(const unsigned short* __restrict__ Q,
                       const unsigned short* __restrict__ Kmat,
                       const unsigned short* __restrict__ Vt,
                       unsigned short* __restrict__ O) {
  constexpr int B = 2, S = 2048, H = 16, Dh = 128;
  constexpr int RS = H * Dh;            // 2048
  const int wave = threadIdx.x >> 5;
  const int lane = threadIdx.x & 31;
  const int lr   = lane & 15;
  const int hi   = lane >> 4;

  constexpr int qtiles = S / 16;
  const int gw = blockIdx.x * FA_WAVES + wave;
  if (gw >= B * H * qtiles) return;
  const int qt = gw % qtiles;
  const int h  = (gw / qtiles) % H;
  const int b  = gw / (qtiles * H);

  __shared__ __attribute__((aligned(16))) unsigned short kst[FA_WAVES][2][32 * Dh];
  __shared__ __attribute__((aligned(16))) unsigned short pbuf[FA_WAVES][16 * 32];
  unsigned short* pb = pbuf[wave];

  const unsigned short* kg = Kmat + ((size_t)b * S * H + h) * Dh;
  const __bf16* qbase = (const __bf16*)Q + ((size_t)b * S * H + h) * Dh;
  const __bf16* vtb   = (const __bf16*)Vt + (((size_t)b * H + h) * Dh) * S;

  // Q fragments: 16 rows x Dh=128 -> 4 chunks of K=32
  V16 qf[4];
  {
    const __bf16* qrow = qbase + (size_t)(qt * 16 + lr) * RS;
#pragma unroll
    for (int t = 0; t < 4; ++t) {
      qf[t].h[0] = *(const v8bf*)(qrow + t * 32 + hi * 8);
      qf[t].h[1] = *(const v8bf*)(qrow + t * 32 + 16 + hi * 8);
    }
  }

  v8f o[8] = {};
  float mrow[8], lrow[8];
#pragma unroll
  for (int i = 0; i < 8; ++i) { mrow[i] = -1e30f; lrow[i] = 0.0f; }

  const int nfull = (qt * 16) / 32;     // fully-unmasked 32-key tiles
  const int ntt   = nfull + 1;          // + exactly one diagonal (masked) tile

  // prologue: fetch key tile 0
  tdm_load_2d_bf16((unsigned)(uintptr_t)&kst[wave][0][0], kg,
                   Dh, 32, RS, Dh, 1u << 20);

  for (int kt = 0; kt < ntt; ++kt) {
    if (kt + 1 < ntt) {
      tdm_load_2d_bf16((unsigned)(uintptr_t)&kst[wave][(kt + 1) & 1][0],
                       kg + (size_t)(kt + 1) * 32 * RS, Dh, 32, RS, Dh, 1u << 20);
      __builtin_amdgcn_s_wait_tensorcnt(1);     // tile kt complete
    } else {
      __builtin_amdgcn_s_wait_tensorcnt(0);
    }
    asm volatile("" ::: "memory");
    const __bf16* ks = (const __bf16*)&kst[wave][kt & 1][0];
    if (kt < nfull)
      fa_process_tile<false>(ks, vtb, pb, qf, o, mrow, lrow, kt * 32, qt, lr, hi);
    else
      fa_process_tile<true>(ks, vtb, pb, qf, o, mrow, lrow, kt * 32, qt, lr, hi);
  }

  // ---- normalize & store head_out[b, q, h, dh] (bf16) ----
  unsigned short* ob = O + ((size_t)b * S * H + h) * Dh;
#pragma unroll
  for (int nt = 0; nt < 8; ++nt) {
#pragma unroll
    for (int i = 0; i < 8; ++i) {
      const int m = qt * 16 + i + 8 * hi;
      const int n = nt * 16 + lr;
      ob[(size_t)m * RS + n] = f2bf_bits(o[nt][i] / lrow[i]);
    }
  }
}

// ---------------------------------------------------------------------------
// Launcher
// ---------------------------------------------------------------------------
extern "C" void kernel_launch(void* const* d_in, const int* in_sizes, int n_in,
                              void* d_out, int out_size, void* d_ws, size_t ws_size,
                              hipStream_t stream) {
  const int B = 2, S = 2048, D = 2048, H = 16, Dh = 128;
  const size_t ND = (size_t)B * S * D;  // 8388608
  const size_t NW = (size_t)D * D;      // 4194304

  const float* x  = (const float*)d_in[0];
  const float* wq = (const float*)d_in[1];
  const float* wk = (const float*)d_in[2];
  const float* wv = (const float*)d_in[3];
  const float* wo = (const float*)d_in[4];
  float* out = (float*)d_out;

  unsigned short* p = (unsigned short*)d_ws;
  unsigned short* xb  = p; p += ND;
  unsigned short* wqb = p; p += NW;
  unsigned short* wkb = p; p += NW;
  unsigned short* wvb = p; p += NW;
  unsigned short* wob = p; p += NW;
  unsigned short* qb  = p; p += ND;
  unsigned short* kb  = p; p += ND;
  unsigned short* vb  = p; p += ND;
  unsigned short* vt  = xb;  // alias: x_bf16 dead after projections
  unsigned short* ho  = vb;  // alias: v_bf16 dead after transpose

  const int thr = 256;

  f32_to_bf16_kernel<<<(int)((ND + thr - 1) / thr), thr, 0, stream>>>(x,  xb,  (int)ND);
  f32_to_bf16_kernel<<<(int)((NW + thr - 1) / thr), thr, 0, stream>>>(wq, wqb, (int)NW);
  f32_to_bf16_kernel<<<(int)((NW + thr - 1) / thr), thr, 0, stream>>>(wk, wkb, (int)NW);
  f32_to_bf16_kernel<<<(int)((NW + thr - 1) / thr), thr, 0, stream>>>(wv, wvb, (int)NW);
  f32_to_bf16_kernel<<<(int)((NW + thr - 1) / thr), thr, 0, stream>>>(wo, wob, (int)NW);

  dim3 ggrid((B * S) / 32, D / 512);
  gemm_xwT_kernel<false><<<ggrid, 256, 0, stream>>>(xb, wqb, qb, nullptr);
  gemm_xwT_kernel<false><<<ggrid, 256, 0, stream>>>(xb, wkb, kb, nullptr);
  gemm_xwT_kernel<false><<<ggrid, 256, 0, stream>>>(xb, wvb, vb, nullptr);

  const int nr = B * S * H * (Dh / 2);
  rope_bf16_kernel<<<(nr + thr - 1) / thr, thr, 0, stream>>>(qb, nr);
  rope_bf16_kernel<<<(nr + thr - 1) / thr, thr, 0, stream>>>(kb, nr);

  transpose_v_kernel<<<(int)((ND + thr - 1) / thr), thr, 0, stream>>>(vb, vt, (int)ND);

  const int total_waves = B * H * (S / 16);
  flash_attn_kernel<<<total_waves / FA_WAVES, FA_WAVES * 32, 0, stream>>>(qb, kb, vt, ho);

  gemm_xwT_kernel<true><<<ggrid, 256, 0, stream>>>(ho, wob, nullptr, out);
}
